// MetaRLScreener_pro_21277267984757
// MI455X (gfx1250) — matrix operand order repaired and checked by reading, not compiled, not run on hardware
//
#include <hip/hip_runtime.h>
#include <hip/hip_bf16.h>
#include <math.h>

typedef __attribute__((ext_vector_type(16))) __bf16 v16bf;
typedef __attribute__((ext_vector_type(8)))  float  v8f;

#define NEG_INF_F (-10000000000.0f)
#define INV_TEMP  (2.0f)   // 1 / 0.5

__device__ __forceinline__ void atomicMaxFloat(float* addr, float val) {
    // sign-aware int punning: works for mixed-sign floats when addr starts at -inf
    if (val >= 0.0f) atomicMax((int*)addr, __float_as_int(val));
    else             atomicMin((unsigned int*)addr, __float_as_uint(val));
}

__global__ void init_kernel(float* ws) {
    ws[0] = __int_as_float(0xFF800000u); // -inf (running max)
    ws[1] = 0.0f;                        // exp-sum
}

// Wave-per-16-edge-tile WMMA kernel: scores -> d_out, global max -> ws[0]
__global__ __launch_bounds__(256) void score_kernel(
    const float* __restrict__ node_reps,   // [N,32]
    const float* __restrict__ edge_reps,   // [E,32]
    const float* __restrict__ graph_rep,   // [32]
    const float* __restrict__ subgraph_rep,// [32]
    const float* __restrict__ W1,          // [96,32] row-major
    const float* __restrict__ b1,          // [32]
    const long long* __restrict__ eidx,    // [2,E] int64
    const unsigned char* __restrict__ sel, // [E] bool
    float* __restrict__ scores,            // [E] (= d_out)
    float* __restrict__ ws,
    long long E)
{
    const int lane = threadIdx.x & 31;
    const int wave = threadIdx.x >> 5;
    const int half = lane >> 4;   // 0: lanes 0-15, 1: lanes 16-31
    const int l15  = lane & 15;

    // Per-lane column scalars for the output dot (C/D layout: n = l15 [+16 for tile1])
    const float bn0 = b1[l15];
    const float bn1 = b1[16 + l15];
    const float gn0 = graph_rep[l15]      - subgraph_rep[l15];
    const float gn1 = graph_rep[16 + l15] - subgraph_rep[16 + l15];

    // Preload B = W1 as six 32x16 bf16 tiles (K-chunk c, N-tile t), register resident.
    // B layout (16-bit, KxN=32x16, wave32): lane l15 -> col n; lanes 0-15 hold K {0..7,16..23},
    // lanes 16-31 hold K {8..15,24..31}, matching the A layout.
    v16bf B[3][2];
    {
        const int kb = half << 3; // 0 or 8
        #pragma unroll
        for (int c = 0; c < 3; ++c) {
            #pragma unroll
            for (int t = 0; t < 2; ++t) {
                const int n = (t << 4) + l15;
                #pragma unroll
                for (int e = 0; e < 16; ++e)
                    B[c][t][e] = (__bf16)W1[(32 * c + kb + e + (e & 8)) * 32 + n];
            }
        }
    }

    const long long waveGlobal = (long long)blockIdx.x * 8 + wave;
    const long long stride     = (long long)gridDim.x * 8 * 16;

    float lmax = __int_as_float(0xFF800000u);

    for (long long base = waveGlobal * 16; base < E; base += stride) {
        // ---- Build A (16 edges x 96 feats, bf16), per the 16-bit 16x32 A layout:
        // lane m covers K {0..7,16..23} of row m; lane m+16 covers K {8..15,24..31}.
        long long er = base + l15;
        if (er >= E) er = E - 1;            // clamp (E is a multiple of 16 anyway)
        const long long si = eidx[er];
        const long long di = eidx[E + er];
        const float* rps0 = node_reps + si * 32;
        const float* rps1 = node_reps + di * 32;
        const float* rps2 = edge_reps + er * 32;
        const int kb = half << 3;           // 0 or 8

        v16bf A[3];
        const float* rps[3] = { rps0, rps1, rps2 };
        #pragma unroll
        for (int c = 0; c < 3; ++c) {
            const float4 f0 = *(const float4*)(rps[c] + kb);       // K kb+0..3
            const float4 f1 = *(const float4*)(rps[c] + kb + 4);   // K kb+4..7
            const float4 f2 = *(const float4*)(rps[c] + kb + 16);  // K kb+16..19
            const float4 f3 = *(const float4*)(rps[c] + kb + 20);  // K kb+20..23
            A[c][0]  = (__bf16)f0.x;  A[c][1]  = (__bf16)f0.y;
            A[c][2]  = (__bf16)f0.z;  A[c][3]  = (__bf16)f0.w;
            A[c][4]  = (__bf16)f1.x;  A[c][5]  = (__bf16)f1.y;
            A[c][6]  = (__bf16)f1.z;  A[c][7]  = (__bf16)f1.w;
            A[c][8]  = (__bf16)f2.x;  A[c][9]  = (__bf16)f2.y;
            A[c][10] = (__bf16)f2.z;  A[c][11] = (__bf16)f2.w;
            A[c][12] = (__bf16)f3.x;  A[c][13] = (__bf16)f3.y;
            A[c][14] = (__bf16)f3.z;  A[c][15] = (__bf16)f3.w;
        }

        // ---- h[16,32] = A @ W1 via 6 WMMAs (2 N-tiles x 3 K-chunks)
        v8f acc0 = {};
        v8f acc1 = {};
        #pragma unroll
        for (int c = 0; c < 3; ++c) {
            acc0 = __builtin_amdgcn_wmma_f32_16x16x32_bf16(
                false, A[c], false, B[c][0], (short)0, acc0, false, false);
            acc1 = __builtin_amdgcn_wmma_f32_16x16x32_bf16(
                false, A[c], false, B[c][1], (short)0, acc1, false, false);
        }

        // ---- bias + ELU + dot with graph_diff, reduce over n (16 lanes per half)
        #pragma unroll
        for (int v = 0; v < 8; ++v) {
            float a0 = acc0[v] + bn0;
            a0 = a0 > 0.0f ? a0 : (__expf(a0) - 1.0f);
            float a1 = acc1[v] + bn1;
            a1 = a1 > 0.0f ? a1 : (__expf(a1) - 1.0f);
            float val = a0 * gn0 + a1 * gn1;
            val += __shfl_xor(val, 8, 16);
            val += __shfl_xor(val, 4, 16);
            val += __shfl_xor(val, 2, 16);
            val += __shfl_xor(val, 1, 16);
            if (l15 == 0) {
                const long long e = base + v + (half << 3);
                if (e < E) {
                    float s = sel[e] ? NEG_INF_F : val;
                    s *= INV_TEMP;
                    scores[e] = s;
                    lmax = fmaxf(lmax, s);
                }
            }
        }
    }

    // ---- block max -> global max
    #pragma unroll
    for (int off = 16; off >= 1; off >>= 1)
        lmax = fmaxf(lmax, __shfl_xor(lmax, off, 32));
    __shared__ float smax[8];
    if (lane == 0) smax[wave] = lmax;
    __syncthreads();
    if (threadIdx.x == 0) {
        float m = smax[0];
        #pragma unroll
        for (int i = 1; i < 8; ++i) m = fmaxf(m, smax[i]);
        atomicMaxFloat(&ws[0], m);
    }
}

// out[i] = exp(out[i] - max); accumulate global sum into ws[1]
__global__ __launch_bounds__(256) void exp_kernel(float* __restrict__ out,
                                                  float* __restrict__ ws,
                                                  long long E) {
    const float gmax = ws[0];
    float lsum = 0.0f;
    for (long long i = (long long)blockIdx.x * 256 + threadIdx.x; i < E;
         i += (long long)gridDim.x * 256) {
        const float e = __expf(out[i] - gmax);
        out[i] = e;
        lsum += e;
    }
    #pragma unroll
    for (int off = 16; off >= 1; off >>= 1)
        lsum += __shfl_xor(lsum, off, 32);
    __shared__ float ssum[8];
    if ((threadIdx.x & 31) == 0) ssum[threadIdx.x >> 5] = lsum;
    __syncthreads();
    if (threadIdx.x == 0) {
        float s = 0.0f;
        #pragma unroll
        for (int i = 0; i < 8; ++i) s += ssum[i];
        atomicAdd(&ws[1], s);
    }
}

// out[i] /= sum
__global__ __launch_bounds__(256) void norm_kernel(float* __restrict__ out,
                                                   const float* __restrict__ ws,
                                                   long long E) {
    const float inv = 1.0f / ws[1];
    for (long long i = (long long)blockIdx.x * 256 + threadIdx.x; i < E;
         i += (long long)gridDim.x * 256)
        out[i] *= inv;
}

extern "C" void kernel_launch(void* const* d_in, const int* in_sizes, int n_in,
                              void* d_out, int out_size, void* d_ws, size_t ws_size,
                              hipStream_t stream) {
    const float*         node_reps    = (const float*)d_in[0];
    const float*         edge_reps    = (const float*)d_in[1];
    const float*         graph_rep    = (const float*)d_in[2];
    const float*         subgraph_rep = (const float*)d_in[3];
    const float*         W1           = (const float*)d_in[4];
    const float*         b1           = (const float*)d_in[5];
    const long long*     eidx         = (const long long*)d_in[6];
    const unsigned char* sel          = (const unsigned char*)d_in[7];

    float* out = (float*)d_out;
    float* ws  = (float*)d_ws;

    const long long E = (long long)in_sizes[7]; // selection: [E]

    init_kernel<<<1, 1, 0, stream>>>(ws);
    score_kernel<<<1024, 256, 0, stream>>>(node_reps, edge_reps, graph_rep,
                                           subgraph_rep, W1, b1, eidx, sel,
                                           out, ws, E);
    exp_kernel<<<2048, 256, 0, stream>>>(out, ws, E);
    norm_kernel<<<2048, 256, 0, stream>>>(out, ws, E);
}
